// GCN_82978768159013
// MI455X (gfx1250) — compile-verified, matrix operand order checked by reading, not run on hardware
//
#include <hip/hip_runtime.h>
#include <hip/hip_bf16.h>

typedef __attribute__((ext_vector_type(16))) __bf16 v16bf;
typedef __attribute__((ext_vector_type(8)))  float  v8f;

#define NN 100000
#define NE 1600000
#define NG 64
#define M_TOT (NE + NN)          // 1,700,000 (divisible by 16)
#define T_MSG (M_TOT / 16)       // 106,250 tiles
#define T_APP (NN / 16)          // 6,250 tiles
#define LN_EPS 1e-5f

// ---------------- zero scratch ----------------
__global__ void zero_kernel(float* p, long n) {
    long i = (long)blockIdx.x * blockDim.x + threadIdx.x;
    if (i < n) p[i] = 0.0f;
}

// ---------------- edge_attr scatter: attr_sum + in_cnt ----------------
__global__ void scatter_attr_kernel(const float* __restrict__ edge_attr,
                                    const int* __restrict__ coli,
                                    float* __restrict__ attr_sum,
                                    float* __restrict__ in_cnt) {
    long tid = (long)blockIdx.x * blockDim.x + threadIdx.x;
    if (tid >= (long)NE * 32) return;
    int e = (int)(tid >> 5);
    int d = (int)(tid & 31);
    int c = coli[e];
    atomicAdd(attr_sum + (size_t)c * 32 + d, edge_attr[(size_t)e * 32 + d]);
    if (d == 0) atomicAdd(in_cnt + c, 1.0f);
}

// ---------------- per node: loop_attr (in place) + norm ----------------
__global__ void node_prep_kernel(float* __restrict__ attr_sum,  // becomes loop_attr
                                 const float* __restrict__ in_cnt,
                                 float* __restrict__ normv) {
    long tid = (long)blockIdx.x * blockDim.x + threadIdx.x;
    if (tid >= (long)NN * 32) return;
    int i = (int)(tid >> 5);
    int d = (int)(tid & 31);
    float cnt = in_cnt[i];
    float v = attr_sum[(size_t)i * 32 + d];
    attr_sum[(size_t)i * 32 + d] = (cnt > 0.0f) ? (v / cnt) : 0.0f;
    if (d == 0) normv[i] = rsqrtf(cnt + 1.0f);   // deg includes self loop
}

// ---------------- pack weight matrix into bf16 WMMA B-fragment layout ----------------
// B fragment for v_wmma_f32_16x16x32_bf16: lane n = lane&15, kb = (lane>>4)*8,
// element i -> k = kc*32 + kb + (i<8 ? i : 8+i)
__global__ void pack_w_kernel(const float* __restrict__ W, __bf16* __restrict__ out,
                              int Kin, int total) {
    int tid = blockIdx.x * blockDim.x + threadIdx.x;
    if (tid >= total) return;
    int i    = tid & 15;
    int lane = (tid >> 4) & 31;
    int frag = tid >> 9;                 // kc*NT + nt  (NT=8)
    int nt = frag & 7;
    int kc = frag >> 3;
    int n  = nt * 16 + (lane & 15);
    int kb = (lane >> 4) << 3;
    int k  = kc * 32 + kb + (i < 8 ? i : 8 + i);
    out[(size_t)frag * 512 + lane * 16 + i] = (__bf16)W[(size_t)n * Kin + k];
}

__device__ __forceinline__ v16bf load_a_frag(const float* src) {
    float4 f0 = *(const float4*)(src + 0);
    float4 f1 = *(const float4*)(src + 4);
    float4 f2 = *(const float4*)(src + 16);
    float4 f3 = *(const float4*)(src + 20);
    v16bf a;
    a[0]=(__bf16)f0.x;  a[1]=(__bf16)f0.y;  a[2]=(__bf16)f0.z;  a[3]=(__bf16)f0.w;
    a[4]=(__bf16)f1.x;  a[5]=(__bf16)f1.y;  a[6]=(__bf16)f1.z;  a[7]=(__bf16)f1.w;
    a[8]=(__bf16)f2.x;  a[9]=(__bf16)f2.y;  a[10]=(__bf16)f2.z; a[11]=(__bf16)f2.w;
    a[12]=(__bf16)f3.x; a[13]=(__bf16)f3.y; a[14]=(__bf16)f3.z; a[15]=(__bf16)f3.w;
    return a;
}

// ---------------- message GEMM: msg = ew * relu([x_j;attr] @ Wmsg^T + b) scatter-> aggr ----------------
__global__ __launch_bounds__(128) void msg_kernel(
    const float* __restrict__ x, const float* __restrict__ edge_attr,
    const float* __restrict__ loop_attr, const int* __restrict__ rowi,
    const int* __restrict__ coli, const float* __restrict__ normv,
    const float* __restrict__ b_msg, const __bf16* __restrict__ Wp,
    float* __restrict__ aggr)
{
    int tile = blockIdx.x * 4 + (threadIdx.x >> 5);
    if (tile >= T_MSG) return;
    int lane = threadIdx.x & 31;
    int mr = lane & 15;
    int kb = (lane >> 4) << 3;
    int e0 = tile * 16;
    int e  = e0 + mr;
    int row = (e < NE) ? rowi[e] : (e - NE);

    v8f acc[8];
    #pragma unroll
    for (int t = 0; t < 8; ++t) {
        #pragma unroll
        for (int j = 0; j < 8; ++j) acc[t][j] = 0.0f;
    }
    const v16bf* Bp = (const v16bf*)Wp;

    #pragma unroll
    for (int kc = 0; kc < 5; ++kc) {
        const float* src;
        if (kc < 4) src = x + (size_t)row * 128 + kc * 32 + kb;
        else        src = ((e < NE) ? (edge_attr + (size_t)e * 32)
                                    : (loop_attr + (size_t)(e - NE) * 32)) + kb;
        v16bf a = load_a_frag(src);
        #pragma unroll
        for (int nt = 0; nt < 8; ++nt) {
            v16bf b = Bp[(kc * 8 + nt) * 32 + lane];
            acc[nt] = __builtin_amdgcn_wmma_f32_16x16x32_bf16(
                false, a, false, b, (short)0, acc[nt], false, false);
        }
    }

    // rows handled by this lane in C/D: m = (lane>>4)*8 + v, col n = nt*16 + (lane&15)
    float wv[8]; int cv[8];
    int mbase = (lane >> 4) << 3;
    #pragma unroll
    for (int v = 0; v < 8; ++v) {
        int em = e0 + mbase + v;
        int r2, c2;
        if (em < NE) { r2 = rowi[em]; c2 = coli[em]; }
        else         { r2 = c2 = em - NE; }
        wv[v] = normv[r2] * normv[c2];
        cv[v] = c2;
    }
    int nlo = lane & 15;
    #pragma unroll
    for (int nt = 0; nt < 8; ++nt) {
        int n = nt * 16 + nlo;
        float bias = b_msg[n];
        #pragma unroll
        for (int v = 0; v < 8; ++v) {
            float val = fmaxf(acc[nt][v] + bias, 0.0f) * wv[v];
            atomicAdd(aggr + (size_t)cv[v] * 128 + n, val);
        }
    }
}

// ---------------- apply GEMM: h = relu([x;aggr] @ Wapply^T + b), fused pool -> gsum ----------------
__global__ __launch_bounds__(128) void apply_kernel(
    const float* __restrict__ x, const float* __restrict__ aggr,
    const int* __restrict__ batch, const float* __restrict__ b_apply,
    const __bf16* __restrict__ Wp, float* __restrict__ gsum)
{
    int tile = blockIdx.x * 4 + (threadIdx.x >> 5);
    if (tile >= T_APP) return;
    int lane = threadIdx.x & 31;
    int node = tile * 16 + (lane & 15);
    int kb = (lane >> 4) << 3;

    v8f acc[8];
    #pragma unroll
    for (int t = 0; t < 8; ++t) {
        #pragma unroll
        for (int j = 0; j < 8; ++j) acc[t][j] = 0.0f;
    }
    const v16bf* Bp = (const v16bf*)Wp;

    #pragma unroll
    for (int kc = 0; kc < 8; ++kc) {
        const float* src = (kc < 4)
            ? (x    + (size_t)node * 128 + kc * 32 + kb)
            : (aggr + (size_t)node * 128 + (kc - 4) * 32 + kb);
        v16bf a = load_a_frag(src);
        #pragma unroll
        for (int nt = 0; nt < 8; ++nt) {
            v16bf b = Bp[(kc * 8 + nt) * 32 + lane];
            acc[nt] = __builtin_amdgcn_wmma_f32_16x16x32_bf16(
                false, a, false, b, (short)0, acc[nt], false, false);
        }
    }

    int bg[8];
    int mbase = (lane >> 4) << 3;
    #pragma unroll
    for (int v = 0; v < 8; ++v) bg[v] = batch[tile * 16 + mbase + v];
    int nlo = lane & 15;
    #pragma unroll
    for (int nt = 0; nt < 8; ++nt) {
        int n = nt * 16 + nlo;
        float bias = b_apply[n];
        #pragma unroll
        for (int v = 0; v < 8; ++v) {
            float val = fmaxf(acc[nt][v] + bias, 0.0f);
            atomicAdd(gsum + (size_t)bg[v] * 128 + n, val);
        }
    }
}

// ---------------- graph node counts ----------------
__global__ void gcnt_kernel(const int* __restrict__ batch, float* __restrict__ gcnt) {
    int i = blockIdx.x * blockDim.x + threadIdx.x;
    if (i >= NN) return;
    atomicAdd(gcnt + batch[i], 1.0f);
}

// ---------------- mean-pool divide + LayerNorm + pred GEMM + relu ----------------
__global__ __launch_bounds__(128) void final_kernel(
    const float* __restrict__ gsum, const float* __restrict__ gcnt,
    const float* __restrict__ gamma, const float* __restrict__ beta,
    const float* __restrict__ Wpred, const float* __restrict__ bpred,
    float* __restrict__ out)
{
    __shared__ float buf[128];
    __shared__ float gn[128];
    int g = blockIdx.x, t = threadIdx.x;
    float cnt = fmaxf(gcnt[g], 1.0f);
    float val = gsum[(size_t)g * 128 + t] / cnt;
    buf[t] = val;
    __syncthreads();
    float s = 0.0f, s2 = 0.0f;
    for (int k = 0; k < 128; ++k) { float u = buf[k]; s += u; s2 += u * u; }
    float mu  = s * (1.0f / 128.0f);
    float var = s2 * (1.0f / 128.0f) - mu * mu;
    float inv = rsqrtf(var + LN_EPS);
    gn[t] = (val - mu) * inv * gamma[t] + beta[t];
    __syncthreads();
    float acc = 0.0f;
    for (int k = 0; k < 128; ++k) acc += gn[k] * Wpred[(size_t)t * 128 + k];
    out[(size_t)g * 128 + t] = fmaxf(acc + bpred[t], 0.0f);
}

extern "C" void kernel_launch(void* const* d_in, const int* in_sizes, int n_in,
                              void* d_out, int out_size, void* d_ws, size_t ws_size,
                              hipStream_t stream) {
    const float* x         = (const float*)d_in[0];
    const float* edge_attr = (const float*)d_in[1];
    const int*   rowi      = (const int*)d_in[2];
    const int*   coli      = rowi + NE;
    const int*   batch     = (const int*)d_in[3];
    const float* W_msg     = (const float*)d_in[4];
    const float* b_msg     = (const float*)d_in[5];
    const float* W_apply   = (const float*)d_in[6];
    const float* b_apply   = (const float*)d_in[7];
    const float* ln_gamma  = (const float*)d_in[8];
    const float* ln_beta   = (const float*)d_in[9];
    const float* W_pred    = (const float*)d_in[10];
    const float* b_pred    = (const float*)d_in[11];
    float* out = (float*)d_out;

    // workspace layout (all 32B aligned)
    float* attr_sum = (float*)d_ws;                  // N*32, becomes loop_attr in place
    float* in_cnt   = attr_sum + (size_t)NN * 32;    // N
    float* normv    = in_cnt + NN;                   // N
    float* aggr     = normv + NN;                    // N*128
    float* gsum     = aggr + (size_t)NN * 128;       // G*128
    float* gcnt     = gsum + (size_t)NG * 128;       // G
    __bf16* pmsg    = (__bf16*)(gcnt + NG);          // 5*8*512 bf16
    __bf16* papp    = pmsg + 5 * 8 * 512;            // 8*8*512 bf16

    long zcount = (long)NN * (32 + 1 + 1 + 128) + (long)NG * 129;
    zero_kernel<<<(int)((zcount + 255) / 256), 256, 0, stream>>>(attr_sum, zcount);

    scatter_attr_kernel<<<(int)(((long)NE * 32 + 255) / 256), 256, 0, stream>>>(
        edge_attr, coli, attr_sum, in_cnt);

    node_prep_kernel<<<(int)(((long)NN * 32 + 255) / 256), 256, 0, stream>>>(
        attr_sum, in_cnt, normv);

    pack_w_kernel<<<(5 * 8 * 512 + 255) / 256, 256, 0, stream>>>(W_msg, pmsg, 160, 5 * 8 * 512);
    pack_w_kernel<<<(8 * 8 * 512 + 255) / 256, 256, 0, stream>>>(W_apply, papp, 256, 8 * 8 * 512);

    msg_kernel<<<(T_MSG + 3) / 4, 128, 0, stream>>>(
        x, edge_attr, attr_sum /*loop_attr*/, rowi, coli, normv, b_msg, pmsg, aggr);

    gcnt_kernel<<<(NN + 255) / 256, 256, 0, stream>>>(batch, gcnt);

    apply_kernel<<<(T_APP + 3) / 4, 128, 0, stream>>>(
        x, aggr, batch, b_apply, papp, gsum);

    final_kernel<<<NG, 128, 0, stream>>>(
        gsum, gcnt, ln_gamma, ln_beta, W_pred, b_pred, out);
}
